// EncDecAD_51840255262852
// MI455X (gfx1250) — matrix operand order repaired and checked by reading, not compiled, not run on hardware
//
#include <hip/hip_runtime.h>
#include <hip/hip_bf16.h>
#include <stdint.h>

// Problem dims
#define Hdim 256
#define Bdim 512
#define Tdim 256
#define GH   1024            // 4*H gate width
#define ROWS 16              // batch rows per workgroup (one WMMA M-tile)
#define KT   32              // K per bf16 WMMA
#define NKT  (Hdim / KT)     // 8 K-tiles per H
#define NBLK (Bdim / ROWS)   // 32 persistent workgroups
#define THREADS 512          // 16 waves -> 4 gate-tiles per wave

typedef __attribute__((ext_vector_type(16))) __bf16 v16bf;
typedef __attribute__((ext_vector_type(8)))  float  v8f;

union Frag32B { uint4 q[2]; v16bf v; };

// ---- WMMA fragment loaders from precomputed per-lane pointers ----
// A-fragment 16x32 bf16: per-lane K groups [0..7] and [16..23] (lane<16), +8 for hi lanes.
__device__ __forceinline__ v16bf fragA_at(const __bf16* p) {
    Frag32B f;
    f.q[0] = *(const uint4*)(p);
    f.q[1] = *(const uint4*)(p + 16);
    return f.v;
}
// A-fragment from fp32 source (convert to bf16 on the fly).
__device__ __forceinline__ v16bf fragA_f32_at(const float* p) {
    v16bf out;
#pragma unroll
    for (int i = 0; i < 8; ++i) {
        out[i]     = (__bf16)p[i];
        out[i + 8] = (__bf16)p[i + 16];
    }
    return out;
}
// B-fragment 32x16 bf16 from B^T row-major [n][k]: per-lane K groups [0..7] and [16..23] (lane<16), +8 for hi lanes.
__device__ __forceinline__ v16bf fragB_at(const __bf16* p) {
    Frag32B f;
    f.q[0] = *(const uint4*)(p);
    f.q[1] = *(const uint4*)(p + 16);
    return f.v;
}

// Branch-free activations on the recurrence critical path (pure v_exp/v_rcp,
// no divergent libm slow path; saturation flows through inf/0 naturally).
__device__ __forceinline__ float sigf(float x) {
    return 1.0f / (1.0f + __expf(-x));
}
__device__ __forceinline__ float tanh_fast(float x) {
    float ex = __expf(2.0f * x);          // x>>0: inf -> 1 ; x<<0: 0 -> -1
    return 1.0f - 2.0f / (ex + 1.0f);
}

// ---- Kernel 0: weight prep (fp32 -> bf16, fuse decoder weights & biases) ----
__global__ __launch_bounds__(256) void prep_kernel(
    const float* __restrict__ Wih1, const float* __restrict__ Whh1,
    const float* __restrict__ bih1, const float* __restrict__ bhh1,
    const float* __restrict__ W1,   const float* __restrict__ b1,
    const float* __restrict__ Wih2, const float* __restrict__ Whh2,
    const float* __restrict__ bih2, const float* __restrict__ bhh2,
    __bf16* __restrict__ wih1b, __bf16* __restrict__ whh1b,
    __bf16* __restrict__ wih2b, __bf16* __restrict__ whh2b,
    __bf16* __restrict__ wcombb, __bf16* __restrict__ w1b,
    float* __restrict__ b1c, float* __restrict__ b2c, float* __restrict__ b1cpy)
{
    const int i = blockIdx.x * blockDim.x + threadIdx.x;
    if (i < GH * Hdim) {
        wih1b[i]  = (__bf16)Wih1[i];
        whh1b[i]  = (__bf16)Whh1[i];
        wih2b[i]  = (__bf16)Wih2[i];
        whh2b[i]  = (__bf16)Whh2[i];
        wcombb[i] = (__bf16)(Wih2[i] + Whh2[i]);  // decoder: input == hidden for t>=1
    }
    if (i < Hdim * Hdim) w1b[i] = (__bf16)W1[i];
    if (i < GH)  { b1c[i] = bih1[i] + bhh1[i]; b2c[i] = bih2[i] + bhh2[i]; }
    if (i < Hdim) b1cpy[i] = b1[i];
}

// ---- Kernel 1: encoder LSTM, persistent per 16-row batch tile ----
__global__ __launch_bounds__(THREADS) void enc_kernel(
    const float*  __restrict__ x,     // (B,T,H) fp32
    const __bf16* __restrict__ Wih,   // (4H,H) bf16 row-major [n][k]
    const __bf16* __restrict__ Whh,
    const float*  __restrict__ bias,  // fused 4H
    __bf16* __restrict__ h1_out,      // (B,H) bf16
    float*  __restrict__ c1_out)      // (B,H) fp32
{
    __shared__ __align__(16) __bf16 lds_h[2][ROWS][Hdim];  // double-buffered hidden (16 KB)
    const int lane = threadIdx.x & 31;
    const int w    = threadIdx.x >> 5;           // wave 0..15, owns h-columns [16w..16w+15]
    const int b0   = blockIdx.x * ROWS;
    const int mOff = (lane & 16) >> 1;           // C/D row offset 0/8
    const int nlan = lane & 15;
    const int col  = 16 * w + nlan;              // this lane's h-column

    for (int i = threadIdx.x; i < ROWS * Hdim; i += THREADS)
        ((__bf16*)lds_h[0])[i] = (__bf16)0.0f;   // h_0 = 0

    // per-lane precomputed offsets (loop-invariant)
    const int aoff = nlan * Hdim + mOff;                                  // A-frag base in LDS h
    const size_t xoff = (size_t)(b0 + nlan) * Tdim * Hdim + (size_t)mOff; // A-frag base in x
    int woff[4];
#pragma unroll
    for (int g = 0; g < 4; ++g) woff[g] = (g * Hdim + 16 * w + nlan) * Hdim + ((lane & 16) >> 1);
    float bias_r[4];
#pragma unroll
    for (int g = 0; g < 4; ++g) bias_r[g] = bias[g * Hdim + col];

    v8f c_state;
#pragma unroll
    for (int e = 0; e < 8; ++e) c_state[e] = 0.0f;

    __syncthreads();

    for (int t = 0; t < Tdim; ++t) {
        const int rb = t & 1, wbuf = rb ^ 1;
        v8f acc[4];
#pragma unroll
        for (int g = 0; g < 4; ++g)
#pragma unroll
            for (int e = 0; e < 8; ++e) acc[g][e] = bias_r[g];

        // prefetch next x_t tile into cache (global_prefetch)
        if (t + 1 < Tdim) {
            const float* nx = x + (size_t)(b0 + (threadIdx.x & 15)) * Tdim * Hdim
                                + (size_t)(t + 1) * Hdim + ((threadIdx.x >> 4) * 8);
            __builtin_prefetch(nx, 0, 1);
        }

        // z += x_t @ W_ih^T   (A from global fp32, cvt to bf16; B streamed from L2)
        const float* xt = x + (size_t)t * Hdim + xoff;
#pragma unroll 1
        for (int kt = 0; kt < NKT; ++kt) {
            v16bf a = fragA_f32_at(xt + kt * KT);
#pragma unroll
            for (int g = 0; g < 4; ++g) {
                v16bf bf = fragB_at(Wih + woff[g] + kt * KT);
                acc[g] = __builtin_amdgcn_wmma_f32_16x16x32_bf16(
                    false, a, false, bf, (short)0, acc[g], false, false);
            }
        }
        // z += h_{t-1} @ W_hh^T   (A from LDS)
        const __bf16* hb = &lds_h[rb][0][0] + aoff;
#pragma unroll 1
        for (int kt = 0; kt < NKT; ++kt) {
            v16bf a = fragA_at(hb + kt * KT);
#pragma unroll
            for (int g = 0; g < 4; ++g) {
                v16bf bf = fragB_at(Whh + woff[g] + kt * KT);
                acc[g] = __builtin_amdgcn_wmma_f32_16x16x32_bf16(
                    false, a, false, bf, (short)0, acc[g], false, false);
            }
        }
        // gates + state update, all within this wave's 16 columns
#pragma unroll
        for (int e = 0; e < 8; ++e) {
            float iv = sigf(acc[0][e]);
            float fv = sigf(acc[1][e]);
            float gv = tanh_fast(acc[2][e]);
            float ov = sigf(acc[3][e]);
            float cv = fv * c_state[e] + iv * gv;
            c_state[e] = cv;
            float hv = ov * tanh_fast(cv);
            const int m = e + mOff;
            lds_h[wbuf][m][col] = (__bf16)hv;
            if (t == Tdim - 1) {
                h1_out[(size_t)(b0 + m) * Hdim + col] = (__bf16)hv;
                c1_out[(size_t)(b0 + m) * Hdim + col] = cv;
            }
        }
        __syncthreads();
    }
}

// ---- Kernel 2: x_end = h1 @ W1^T + b1  (bf16 out for decoder input) ----
__global__ __launch_bounds__(THREADS) void xend_kernel(
    const __bf16* __restrict__ h1, const __bf16* __restrict__ W1b,
    const float* __restrict__ b1, __bf16* __restrict__ xend)
{
    const int lane = threadIdx.x & 31;
    const int w    = threadIdx.x >> 5;   // wave w owns N-tile w (16 cols)
    const int b0   = blockIdx.x * ROWS;
    const int mOff = (lane & 16) >> 1;
    const int nlan = lane & 15;
    const int col  = 16 * w + nlan;

    const size_t aoff = (size_t)(b0 + nlan) * Hdim + (size_t)mOff;
    const int woff = (16 * w + nlan) * Hdim + ((lane & 16) >> 1);

    v8f acc;
    {
        float bv = b1[col];
#pragma unroll
        for (int e = 0; e < 8; ++e) acc[e] = bv;
    }
#pragma unroll 1
    for (int kt = 0; kt < NKT; ++kt) {
        v16bf a  = fragA_at(h1 + aoff + kt * KT);
        v16bf bf = fragB_at(W1b + woff + kt * KT);
        acc = __builtin_amdgcn_wmma_f32_16x16x32_bf16(
            false, a, false, bf, (short)0, acc, false, false);
    }
#pragma unroll
    for (int e = 0; e < 8; ++e)
        xend[(size_t)(b0 + e + mOff) * Hdim + col] = (__bf16)acc[e];
}

// ---- Kernel 3: decoder LSTM, persistent; writes fp32 output at reversed index ----
__global__ __launch_bounds__(THREADS) void dec_kernel(
    const __bf16* __restrict__ xend, const __bf16* __restrict__ h1,
    const float*  __restrict__ c1,
    const __bf16* __restrict__ Wih2, const __bf16* __restrict__ Whh2,
    const __bf16* __restrict__ Wcomb,   // W_ih2 + W_hh2 (t>=1: input == hidden)
    const float*  __restrict__ bias,
    float* __restrict__ out)            // (B,T,H) fp32
{
    __shared__ __align__(16) __bf16 lds_h[2][ROWS][Hdim];
    const int lane = threadIdx.x & 31;
    const int w    = threadIdx.x >> 5;
    const int b0   = blockIdx.x * ROWS;
    const int mOff = (lane & 16) >> 1;
    const int nlan = lane & 15;
    const int col  = 16 * w + nlan;

    for (int i = threadIdx.x; i < ROWS * Hdim; i += THREADS) {
        int r = i >> 8, k = i & (Hdim - 1);
        lds_h[0][r][k] = h1[(size_t)(b0 + r) * Hdim + k];
    }

    const int aoff = nlan * Hdim + mOff;
    const size_t xeoff = (size_t)(b0 + nlan) * Hdim + (size_t)mOff;
    int woff[4];
#pragma unroll
    for (int g = 0; g < 4; ++g) woff[g] = (g * Hdim + 16 * w + nlan) * Hdim + ((lane & 16) >> 1);
    float bias_r[4];
#pragma unroll
    for (int g = 0; g < 4; ++g) bias_r[g] = bias[g * Hdim + col];

    v8f c_state;
#pragma unroll
    for (int e = 0; e < 8; ++e)
        c_state[e] = c1[(size_t)(b0 + e + mOff) * Hdim + col];

    __syncthreads();

    for (int t = 0; t < Tdim; ++t) {
        const int rb = t & 1, wbuf = rb ^ 1;
        v8f acc[4];
#pragma unroll
        for (int g = 0; g < 4; ++g)
#pragma unroll
            for (int e = 0; e < 8; ++e) acc[g][e] = bias_r[g];

        if (t == 0) {
            // z = x_end @ W_ih2^T + h1 @ W_hh2^T
#pragma unroll 1
            for (int kt = 0; kt < NKT; ++kt) {
                v16bf a = fragA_at(xend + xeoff + kt * KT);
#pragma unroll
                for (int g = 0; g < 4; ++g) {
                    v16bf bf = fragB_at(Wih2 + woff[g] + kt * KT);
                    acc[g] = __builtin_amdgcn_wmma_f32_16x16x32_bf16(
                        false, a, false, bf, (short)0, acc[g], false, false);
                }
            }
            const __bf16* hb = &lds_h[0][0][0] + aoff;
#pragma unroll 1
            for (int kt = 0; kt < NKT; ++kt) {
                v16bf a = fragA_at(hb + kt * KT);
#pragma unroll
                for (int g = 0; g < 4; ++g) {
                    v16bf bf = fragB_at(Whh2 + woff[g] + kt * KT);
                    acc[g] = __builtin_amdgcn_wmma_f32_16x16x32_bf16(
                        false, a, false, bf, (short)0, acc[g], false, false);
                }
            }
        } else {
            // z = h_{t-1} @ (W_ih2 + W_hh2)^T  -- one fused GEMM
            const __bf16* hb = &lds_h[rb][0][0] + aoff;
#pragma unroll 1
            for (int kt = 0; kt < NKT; ++kt) {
                v16bf a = fragA_at(hb + kt * KT);
#pragma unroll
                for (int g = 0; g < 4; ++g) {
                    v16bf bf = fragB_at(Wcomb + woff[g] + kt * KT);
                    acc[g] = __builtin_amdgcn_wmma_f32_16x16x32_bf16(
                        false, a, false, bf, (short)0, acc[g], false, false);
                }
            }
        }

        const int j = (t < Tdim - 1) ? (Tdim - 2 - t) : (Tdim - 1);  // time-reversed slot
#pragma unroll
        for (int e = 0; e < 8; ++e) {
            float iv = sigf(acc[0][e]);
            float fv = sigf(acc[1][e]);
            float gv = tanh_fast(acc[2][e]);
            float ov = sigf(acc[3][e]);
            float cv = fv * c_state[e] + iv * gv;
            c_state[e] = cv;
            float hv = ov * tanh_fast(cv);
            const int m = e + mOff;
            lds_h[wbuf][m][col] = (__bf16)hv;
            out[(size_t)(b0 + m) * Tdim * Hdim + (size_t)j * Hdim + col] = hv;
        }
        __syncthreads();
    }
}

// ---- Host launcher ----
extern "C" void kernel_launch(void* const* d_in, const int* in_sizes, int n_in,
                              void* d_out, int out_size, void* d_ws, size_t ws_size,
                              hipStream_t stream) {
    (void)in_sizes; (void)n_in; (void)out_size; (void)ws_size;
    const float* x    = (const float*)d_in[0];
    const float* Wih1 = (const float*)d_in[1];
    const float* Whh1 = (const float*)d_in[2];
    const float* bih1 = (const float*)d_in[3];
    const float* bhh1 = (const float*)d_in[4];
    const float* W1   = (const float*)d_in[5];
    const float* b1   = (const float*)d_in[6];
    const float* Wih2 = (const float*)d_in[7];
    const float* Whh2 = (const float*)d_in[8];
    const float* bih2 = (const float*)d_in[9];
    const float* bhh2 = (const float*)d_in[10];
    float* out = (float*)d_out;

    char* ws = (char*)d_ws;
    size_t o = 0;
    auto alloc = [&](size_t bytes) -> char* {
        char* p = ws + o;
        o += (bytes + 255) & ~(size_t)255;
        return p;
    };
    __bf16* wih1b  = (__bf16*)alloc((size_t)GH * Hdim * sizeof(__bf16));
    __bf16* whh1b  = (__bf16*)alloc((size_t)GH * Hdim * sizeof(__bf16));
    __bf16* wih2b  = (__bf16*)alloc((size_t)GH * Hdim * sizeof(__bf16));
    __bf16* whh2b  = (__bf16*)alloc((size_t)GH * Hdim * sizeof(__bf16));
    __bf16* wcombb = (__bf16*)alloc((size_t)GH * Hdim * sizeof(__bf16));
    __bf16* w1b    = (__bf16*)alloc((size_t)Hdim * Hdim * sizeof(__bf16));
    float*  b1c    = (float*)alloc(GH * sizeof(float));
    float*  b2c    = (float*)alloc(GH * sizeof(float));
    float*  b1f    = (float*)alloc(Hdim * sizeof(float));
    __bf16* h1b    = (__bf16*)alloc((size_t)Bdim * Hdim * sizeof(__bf16));
    float*  c1f    = (float*)alloc((size_t)Bdim * Hdim * sizeof(float));
    __bf16* xendb  = (__bf16*)alloc((size_t)Bdim * Hdim * sizeof(__bf16));

    prep_kernel<<<(GH * Hdim) / 256, 256, 0, stream>>>(
        Wih1, Whh1, bih1, bhh1, W1, b1, Wih2, Whh2, bih2, bhh2,
        wih1b, whh1b, wih2b, whh2b, wcombb, w1b, b1c, b2c, b1f);

    enc_kernel<<<NBLK, THREADS, 0, stream>>>(x, wih1b, whh1b, b1c, h1b, c1f);
    xend_kernel<<<NBLK, THREADS, 0, stream>>>(h1b, w1b, b1f, xendb);
    dec_kernel<<<NBLK, THREADS, 0, stream>>>(xendb, h1b, c1f, wih2b, whh2b, wcombb, b2c, out);
}
